// AtomTransformerBlock_34789235098051
// MI455X (gfx1250) — matrix operand order repaired and checked by reading, not compiled, not run on hardware
//
#include <hip/hip_runtime.h>
#include <math.h>

// ---------------------------------------------------------------------------
// AtomTransformer block for MI455X (gfx1250, wave32, WMMA).
// - All GEMMs: v_wmma_f32_16x16x32_bf16, 32x32 tile per wave (4 accumulators,
//   2 A-frags x 2 B-frags -> 4 WMMAs per 8 b128 loads).
// - Q/K/V fused into one 2048x768x256 GEMM (shared A = LN(x)).
// - pair_mean computed only for gathered (n, block_index[n,b]) pairs:
//   33.5 MB of reads instead of 537 MB for the full (N,N,CP) tensor.
// ---------------------------------------------------------------------------

#define NTOK 2048
#define CCH  256
#define HDS  8
#define DH   32
#define CP   32
#define BNB  128
#define QKVS 768   // fused qkv row stride

typedef __bf16 bf16_t;
typedef __attribute__((ext_vector_type(16))) __bf16 v16bf;
typedef __attribute__((ext_vector_type(8)))  float  v8f;

// ---------------------------------------------------------------- convert ---
__global__ __launch_bounds__(256)
void cvt_f32_to_bf16(const float* __restrict__ in, bf16_t* __restrict__ out, int n) {
    int i = blockIdx.x * blockDim.x + threadIdx.x;
    int stride = gridDim.x * blockDim.x;
    for (; i < n; i += stride) out[i] = (bf16_t)in[i];
}

// -------------------------------------------------------------- layernorm ---
// one wave (32 lanes) per row of 256 floats; 8 waves per block
__global__ __launch_bounds__(256)
void layernorm_to_bf16(const float* __restrict__ x, bf16_t* __restrict__ out) {
    const int wave = threadIdx.x >> 5;
    const int lane = threadIdx.x & 31;
    const int row  = blockIdx.x * 8 + wave;
    const float* xr = x + (size_t)row * CCH;

    float4 a = ((const float4*)xr)[lane * 2 + 0];
    float4 b = ((const float4*)xr)[lane * 2 + 1];
    float s = a.x + a.y + a.z + a.w + b.x + b.y + b.z + b.w;
    #pragma unroll
    for (int m = 16; m >= 1; m >>= 1) s += __shfl_xor(s, m, 32);
    const float mean = s * (1.0f / CCH);

    float v = 0.f;
    v += (a.x-mean)*(a.x-mean) + (a.y-mean)*(a.y-mean) + (a.z-mean)*(a.z-mean) + (a.w-mean)*(a.w-mean);
    v += (b.x-mean)*(b.x-mean) + (b.y-mean)*(b.y-mean) + (b.z-mean)*(b.z-mean) + (b.w-mean)*(b.w-mean);
    #pragma unroll
    for (int m = 16; m >= 1; m >>= 1) v += __shfl_xor(v, m, 32);
    const float inv = rsqrtf(v * (1.0f / CCH) + 1e-5f);

    bf16_t* o = out + (size_t)row * CCH + lane * 8;
    o[0] = (bf16_t)((a.x - mean) * inv);
    o[1] = (bf16_t)((a.y - mean) * inv);
    o[2] = (bf16_t)((a.z - mean) * inv);
    o[3] = (bf16_t)((a.w - mean) * inv);
    o[4] = (bf16_t)((b.x - mean) * inv);
    o[5] = (bf16_t)((b.y - mean) * inv);
    o[6] = (bf16_t)((b.z - mean) * inv);
    o[7] = (bf16_t)((b.w - mean) * inv);
}

// ------------------------------------------------------------- WMMA GEMM ----
// out[m, n] = sum_k A[m, k] * Wt[n, k]   (A @ W^T, both row-major bf16)
// 256 threads = 8 waves; each wave owns a 32x32 tile (4 WMMA accumulators);
// waves arranged 2(M) x 4(N) -> block tile 64 x 128.
// EPI: 0 = f32 store, 1 = +resid f32, 2 = +bias,SiLU -> bf16, 3 = +bias+resid f32
template<int EPI>
__global__ __launch_bounds__(256)
void gemm_bf16_wmma(const bf16_t* __restrict__ A,
                    const bf16_t* __restrict__ Wt,
                    const float*  __restrict__ bias,
                    const float*  __restrict__ resid,
                    float*        __restrict__ outF,
                    bf16_t*       __restrict__ outB,
                    int M, int Nout, int K) {
    const int wv   = threadIdx.x >> 5;
    const int lane = threadIdx.x & 31;
    const int r    = lane & 15;
    const int half = lane >> 4;

    const int m_base = blockIdx.x * 64  + (wv & 1) * 32;
    const int n_base = blockIdx.y * 128 + (wv >> 1) * 32;

    const bf16_t* arow[2] = { A  + (size_t)(m_base      + r) * K,
                              A  + (size_t)(m_base + 16 + r) * K };
    const bf16_t* brow[2] = { Wt + (size_t)(n_base      + r) * K,
                              Wt + (size_t)(n_base + 16 + r) * K };

    v8f acc[2][2] = {};
    union Frag { uint4 u[2]; v16bf v; };

    for (int kb = 0; kb < K; kb += 32) {
        Frag fa[2], fb[2];
        #pragma unroll
        for (int i = 0; i < 2; ++i) {
            // A fragment (16x32): per lane K = half*8 + [0..7] and 16+half*8 + [0..7]
            fa[i].u[0] = *(const uint4*)(arow[i] + kb + half * 8);
            fa[i].u[1] = *(const uint4*)(arow[i] + kb + 16 + half * 8);
            // B fragment (32x16): per lane (column = r) K = half*8 + [0..7] and 16+half*8 + [0..7]
            fb[i].u[0] = *(const uint4*)(brow[i] + kb + half * 8);
            fb[i].u[1] = *(const uint4*)(brow[i] + kb + 16 + half * 8);
        }
        #pragma unroll
        for (int mt = 0; mt < 2; ++mt)
            #pragma unroll
            for (int nt = 0; nt < 2; ++nt)
                acc[mt][nt] = __builtin_amdgcn_wmma_f32_16x16x32_bf16(
                    false, fa[mt].v, false, fb[nt].v, (short)0, acc[mt][nt], false, false);
    }

    #pragma unroll
    for (int mt = 0; mt < 2; ++mt) {
        #pragma unroll
        for (int nt = 0; nt < 2; ++nt) {
            const int nn = n_base + nt * 16 + r;
            #pragma unroll
            for (int i = 0; i < 8; ++i) {
                const int m = m_base + mt * 16 + half * 8 + i;
                const size_t idx = (size_t)m * Nout + nn;
                float t = acc[mt][nt][i];
                if (EPI == 0) {
                    outF[idx] = t;
                } else if (EPI == 1) {
                    outF[idx] = resid[idx] + t;
                } else if (EPI == 2) {
                    t += bias[nn];
                    outB[idx] = (bf16_t)(t / (1.0f + __expf(-t)));   // SiLU
                } else { // 3
                    outF[idx] = resid[idx] + t + bias[nn];
                }
            }
        }
    }
}

// -------------------------------------------------------------- attention ---
// One workgroup (256 threads) per token; qkv fused with row stride 768
// (q = cols 0..255, k = 256..511, v = 512..767).
//  phase A: threads (b, half-of-heads) compute gathered pair-bias + q.k scores
//  phase B: softmax over B=128 neighbors in LDS
//  phase C: thread t = (h,d) accumulates sum_b w[h,b] * v[j_b, h, d]
//           (for fixed b the 256 threads read one contiguous v row -> coalesced)
__global__ __launch_bounds__(256)
void attn_gather(const float* __restrict__ qkv,
                 const long long* __restrict__ block_index,
                 const float* __restrict__ pair_emb,
                 const float* __restrict__ Wb,
                 bf16_t* __restrict__ attn_out) {
    const int n = blockIdx.x;
    const int t = threadIdx.x;

    __shared__ float q_s[CCH];
    __shared__ float sc[HDS][BNB + 1];
    __shared__ int   js[BNB];
    __shared__ float m_s[HDS], inv_s[HDS], wb_s[HDS];

    const float scale = 0.17677669529663687f;   // 1/sqrt(32)
    q_s[t] = qkv[(size_t)n * QKVS + t] * scale;
    if (t < HDS) wb_s[t] = Wb[t];
    __syncthreads();

    // ---- phase A: scores -------------------------------------------------
    {
        const int b    = t & (BNB - 1);
        const int hgrp = t >> 7;                 // 0 -> heads 0..3, 1 -> 4..7
        const long long j = block_index[(size_t)n * BNB + b];
        if (hgrp == 0) {
            js[b] = (int)j;
            // warm the gathered V row for phase C (global_prefetch_b8)
            const float* vrow = qkv + (size_t)j * QKVS + 2 * CCH;
            __builtin_prefetch(vrow, 0, 1);
            __builtin_prefetch(vrow + 128, 0, 1);
        }

        // gathered pair mean over CP=32 channels
        const float4* pp = (const float4*)(pair_emb + ((size_t)n * NTOK + (size_t)j) * CP);
        float pm = 0.f;
        #pragma unroll
        for (int c = 0; c < 8; ++c) {
            float4 p = pp[c];
            pm += p.x + p.y + p.z + p.w;
        }
        pm *= (1.0f / CP);

        const int h0 = hgrp * 4;
        #pragma unroll
        for (int h = h0; h < h0 + 4; ++h) {
            const float4* kp = (const float4*)(qkv + (size_t)j * QKVS + CCH + h * DH);
            const float4* qp = (const float4*)(&q_s[h * DH]);
            float s = 0.f;
            #pragma unroll
            for (int d4 = 0; d4 < 8; ++d4) {
                float4 kk = kp[d4];
                float4 qq = qp[d4];
                s += qq.x * kk.x + qq.y * kk.y + qq.z * kk.z + qq.w * kk.w;
            }
            sc[h][b] = s + pm * wb_s[h];
        }
    }
    __syncthreads();

    // ---- phase B: softmax over neighbors ---------------------------------
    if (t < HDS) {
        float m = -3.0e38f;
        for (int b = 0; b < BNB; ++b) m = fmaxf(m, sc[t][b]);
        m_s[t] = m;
    }
    __syncthreads();
    #pragma unroll
    for (int i = 0; i < 4; ++i) {
        const int idx = t * 4 + i;
        const int h = idx >> 7, b = idx & (BNB - 1);
        sc[h][b] = __expf(sc[h][b] - m_s[h]);
    }
    __syncthreads();
    if (t < HDS) {
        float s = 0.f;
        for (int b = 0; b < BNB; ++b) s += sc[t][b];
        inv_s[t] = 1.0f / s;
    }
    __syncthreads();

    // ---- phase C: weighted V aggregation ---------------------------------
    const int h = t >> 5;                        // c = t = h*32 + d
    float acc = 0.f;
    for (int b = 0; b < BNB; ++b) {
        acc += sc[h][b] * qkv[(size_t)js[b] * QKVS + 2 * CCH + t];
    }
    attn_out[(size_t)n * CCH + t] = (bf16_t)(acc * inv_s[h]);
}

// ------------------------------------------------------------------ launch --
extern "C" void kernel_launch(void* const* d_in, const int* in_sizes, int n_in,
                              void* d_out, int out_size, void* d_ws, size_t ws_size,
                              hipStream_t stream) {
    const float*     x        = (const float*)d_in[0];
    const float*     pair_emb = (const float*)d_in[1];
    const long long* blk_idx  = (const long long*)d_in[2];
    const float*     Wq       = (const float*)d_in[3];
    const float*     Wk       = (const float*)d_in[4];
    const float*     Wv       = (const float*)d_in[5];
    const float*     Wb       = (const float*)d_in[6];
    const float*     Wout     = (const float*)d_in[7];
    const float*     W1       = (const float*)d_in[8];
    const float*     b1       = (const float*)d_in[9];
    const float*     W2       = (const float*)d_in[10];
    const float*     b2       = (const float*)d_in[11];
    float*           out      = (float*)d_out;

    // ---- workspace carve-up (aligned to 256 B) ----
    size_t off = 0;
    auto alloc = [&](size_t bytes) {
        void* p = (char*)d_ws + off;
        off += (bytes + 255) & ~(size_t)255;
        return p;
    };
    bf16_t* x_ln_bf   = (bf16_t*)alloc((size_t)NTOK * CCH * 2);
    bf16_t* attn_bf   = (bf16_t*)alloc((size_t)NTOK * CCH * 2);
    bf16_t* h_ln_bf   = (bf16_t*)alloc((size_t)NTOK * CCH * 2);
    bf16_t* hidden_bf = (bf16_t*)alloc((size_t)NTOK * 4 * CCH * 2);
    float*  qkvf      = (float*)alloc((size_t)NTOK * QKVS * 4);
    float*  x_mid     = (float*)alloc((size_t)NTOK * CCH * 4);
    bf16_t* Wqkv_bf   = (bf16_t*)alloc((size_t)3 * CCH * CCH * 2);
    bf16_t* Wout_bf   = (bf16_t*)alloc((size_t)CCH * CCH * 2);
    bf16_t* W1_bf     = (bf16_t*)alloc((size_t)4 * CCH * CCH * 2);
    bf16_t* W2_bf     = (bf16_t*)alloc((size_t)4 * CCH * CCH * 2);
    (void)ws_size; (void)in_sizes; (void)n_in; (void)out_size;

    // 1) weights -> bf16 (Wq/Wk/Wv packed into one 768x256 buffer)
    cvt_f32_to_bf16<<<64, 256, 0, stream>>>(Wq,   Wqkv_bf,                 CCH * CCH);
    cvt_f32_to_bf16<<<64, 256, 0, stream>>>(Wk,   Wqkv_bf + CCH * CCH,     CCH * CCH);
    cvt_f32_to_bf16<<<64, 256, 0, stream>>>(Wv,   Wqkv_bf + 2 * CCH * CCH, CCH * CCH);
    cvt_f32_to_bf16<<<64, 256, 0, stream>>>(Wout, Wout_bf,                 CCH * CCH);
    cvt_f32_to_bf16<<<256, 256, 0, stream>>>(W1,  W1_bf,               4 * CCH * CCH);
    cvt_f32_to_bf16<<<256, 256, 0, stream>>>(W2,  W2_bf,               4 * CCH * CCH);

    // 2) LN1
    layernorm_to_bf16<<<NTOK / 8, 256, 0, stream>>>(x, x_ln_bf);

    // 3) fused QKV GEMM: 2048 x 768 x 256 (bf16 WMMA, f32 out)
    gemm_bf16_wmma<0><<<dim3(NTOK / 64, QKVS / 128), 256, 0, stream>>>(
        x_ln_bf, Wqkv_bf, nullptr, nullptr, qkvf, nullptr, NTOK, QKVS, CCH);

    // 4) gathered attention (pair-mean only for gathered pairs)
    attn_gather<<<NTOK, 256, 0, stream>>>(qkvf, blk_idx, pair_emb, Wb, attn_bf);

    // 5) x_mid = x + attn @ Wout^T
    gemm_bf16_wmma<1><<<dim3(NTOK / 64, CCH / 128), 256, 0, stream>>>(
        attn_bf, Wout_bf, nullptr, x, x_mid, nullptr, NTOK, CCH, CCH);

    // 6) LN2
    layernorm_to_bf16<<<NTOK / 8, 256, 0, stream>>>(x_mid, h_ln_bf);

    // 7) hidden = silu(h_ln @ W1^T + b1)   (2048x1024x256)
    gemm_bf16_wmma<2><<<dim3(NTOK / 64, (4 * CCH) / 128), 256, 0, stream>>>(
        h_ln_bf, W1_bf, b1, nullptr, nullptr, hidden_bf, NTOK, 4 * CCH, CCH);

    // 8) out = x_mid + hidden @ W2^T + b2  (2048x256x1024)
    gemm_bf16_wmma<3><<<dim3(NTOK / 64, CCH / 128), 256, 0, stream>>>(
        hidden_bf, W2_bf, b2, x_mid, out, nullptr, NTOK, CCH, 4 * CCH);
}
